// BatchedExperts_12773232738931
// MI455X (gfx1250) — matrix operand order repaired and checked by reading, not compiled, not run on hardware
//
#include <hip/hip_runtime.h>

// ---------------------------------------------------------------------------
// MoE batched experts for MI455X (gfx1250, wave32, WMMA).
// Route tokens -> per-expert dense GEMMs with split-bf16 (hi/lo) WMMA for
// ~fp32 accuracy at matrix-core rate. With routing (1/8 the reference FLOPs)
// the problem is HBM-bound (~330MB @ 23.3TB/s ~= 14us), so the 3x-WMMA split
// is free at the roofline. Mainloop: ds_load_b128 fragments + v_wmma, with
// global loads software-pipelined (issued one k-step ahead into registers).
// ---------------------------------------------------------------------------

#define N_EXPERTS  8
#define D_MODEL    1024
#define D_FF       2048
#define NUM_TOKENS 16384

#define TM   128      // token rows per block tile
#define TFN  64       // gate/up (GEMM1) or d_model (GEMM2) cols per block tile
#define LDA  40       // LDS row stride in shorts: 80B rows -> 16B aligned,
                      // 20 dwords/row -> 16 distinct banks across 16 lanes

typedef __attribute__((ext_vector_type(16))) __bf16          v16bf;
typedef __attribute__((ext_vector_type(16))) unsigned short  v16us;
typedef __attribute__((ext_vector_type(8)))  float           v8f;

// ---- fp32 <-> bf16 helpers (RNE), split fp32 into bf16 hi + bf16 lo -------
__device__ __forceinline__ unsigned short f2bf(float f) {
    unsigned int u = __float_as_uint(f);
    u += 0x7FFFu + ((u >> 16) & 1u);
    return (unsigned short)(u >> 16);
}
__device__ __forceinline__ float bf2f(unsigned short h) {
    return __uint_as_float(((unsigned int)h) << 16);
}
__device__ __forceinline__ void split_bf16(float f, unsigned short& hi, unsigned short& lo) {
    hi = f2bf(f);
    lo = f2bf(f - bf2f(hi));
}
__device__ __forceinline__ v8f vzero8() {
    v8f z;
#pragma unroll
    for (int i = 0; i < 8; ++i) z[i] = 0.0f;
    return z;
}
// convert one float4 into hi/lo ushort4 and store to LDS (8B-aligned rows)
__device__ __forceinline__ void store_split4(unsigned short* dh, unsigned short* dl,
                                             int off, float4 v) {
    unsigned short h0,h1,h2,h3,l0,l1,l2,l3;
    split_bf16(v.x,h0,l0); split_bf16(v.y,h1,l1);
    split_bf16(v.z,h2,l2); split_bf16(v.w,h3,l3);
    *(ushort4*)&dh[off] = make_ushort4(h0,h1,h2,h3);
    *(ushort4*)&dl[off] = make_ushort4(l0,l1,l2,l3);
}

// ---- WMMA wrapper ---------------------------------------------------------
__device__ __forceinline__ v8f wmma_bf16(v16bf a, v16bf b, v8f c) {
    return __builtin_amdgcn_wmma_f32_16x16x32_bf16(false, a, false, b, (short)0, c, false, false);
}

// A fragment: 16x32 bf16 tile. Lane L(<16): M=m0+L, K=0..15;
// lane L+16: same M, K=16..31.
__device__ __forceinline__ v16bf load_frag_A(const unsigned short* s, int m0, int lane) {
    const int half = lane >> 4;
    const int mr   = lane & 15;
    const unsigned short* p = s + (m0 + mr) * LDA + half * 16;
    v16us t;
#pragma unroll
    for (int i = 0; i < 16; ++i) t[i] = p[i];
    return __builtin_bit_cast(v16bf, t);
}

// B fragment: 32x16 bf16 (K x N), LDS holds W[j][k] rows (j = output col).
// Lane L(<16): N=j0+L, K=0..15; lane L+16: same N, K=16..31.
__device__ __forceinline__ v16bf load_frag_B(const unsigned short* s, int j0, int lane) {
    const int n  = lane & 15;
    const int kb = (lane >> 4) * 16;
    const unsigned short* p = s + (j0 + n) * LDA + kb;
    v16us t;
#pragma unroll
    for (int i = 0; i < 16; ++i) t[i] = p[i];
    return __builtin_bit_cast(v16bf, t);
}

// ---------------------------------------------------------------------------
// Routing kernels
// ---------------------------------------------------------------------------
__global__ void k_init(int* counts, int* cursors) {
    int t = threadIdx.x;
    if (t < N_EXPERTS) { counts[t] = 0; cursors[t] = 0; }
}
__global__ void k_count(const int* __restrict__ ei, int* __restrict__ counts) {
    int t = blockIdx.x * 256 + threadIdx.x;
    if (t < NUM_TOKENS) atomicAdd(&counts[ei[t]], 1);
}
__global__ void k_scan(const int* __restrict__ counts, int* __restrict__ offsets) {
    if (threadIdx.x == 0) {
        int s = 0;
        for (int e = 0; e < N_EXPERTS; ++e) { offsets[e] = s; s += counts[e]; }
        offsets[N_EXPERTS] = s;
    }
}
__global__ void k_scatter(const int* __restrict__ ei, const int* __restrict__ offsets,
                          int* __restrict__ cursors, int* __restrict__ perm) {
    int t = blockIdx.x * 256 + threadIdx.x;
    if (t < NUM_TOKENS) {
        int e = ei[t];
        int p = offsets[e] + atomicAdd(&cursors[e], 1);
        perm[p] = t;
    }
}

// ---------------------------------------------------------------------------
// GEMM1 + SiLU*up:  hidden[p, f] = silu(x·Wg^T) * (x·Wu^T), bf16 hi/lo to ws.
// Block: 256 thr = 8 waves (4Mx2F); each wave: 32x32 gate + 32x32 up.
// ---------------------------------------------------------------------------
__global__ __launch_bounds__(256)
void k_gemm1(const float* __restrict__ x,
             const float* __restrict__ gup,        // [8][2*D_FF][D_MODEL]
             const int* __restrict__ offsets,
             const int* __restrict__ perm,
             unsigned short* __restrict__ hh,
             unsigned short* __restrict__ hl) {
    const int e       = blockIdx.y;
    const int base    = offsets[e];
    const int cnt     = offsets[e + 1] - base;
    const int m_start = blockIdx.x * TM;
    if (m_start >= cnt) return;                    // uniform: whole block exits
    const int f0   = blockIdx.z * TFN;
    const int tid  = threadIdx.x;
    const int lane = tid & 31;
    const int w    = tid >> 5;
    const int wr   = w >> 1;
    const int wc   = w & 1;

    __shared__ unsigned short sAh[TM * LDA], sAl[TM * LDA];
    __shared__ unsigned short sBh[TM * LDA], sBl[TM * LDA]; // rows 0-63 gate, 64-127 up
    __shared__ int sTok[TM];

    if (tid < TM) {
        int r = m_start + tid;
        sTok[tid] = perm[base + (r < cnt ? r : cnt - 1)];
    }
    __syncthreads();

    // ---- per-thread staging pointers, hoisted out of the k-loop -----------
    const float* wptr = gup + (size_t)e * (2u * D_FF) * D_MODEL;
    const float* aptr[4];
    const float* bptr[4];
    int soff[4];
#pragma unroll
    for (int i = 0; i < 4; ++i) {
        int linear = tid + i * 256;                // 0..1023 over float4 slots
        int row = linear >> 3;                     // 0..127
        int kq  = (linear & 7) * 4;                // 0..28
        soff[i] = row * LDA + kq;
        aptr[i] = x + (size_t)sTok[row] * D_MODEL + kq;
        int j   = (row < TFN) ? (f0 + row) : (D_FF + f0 + (row - TFN));
        bptr[i] = wptr + (size_t)j * D_MODEL + kq;
    }

    v8f accg[2][2], accu[2][2];
#pragma unroll
    for (int i = 0; i < 2; ++i)
#pragma unroll
        for (int j = 0; j < 2; ++j) { accg[i][j] = vzero8(); accu[i][j] = vzero8(); }

    // ---- software pipeline: preload k-chunk 0 into registers --------------
    float4 ra[4], rb[4];
#pragma unroll
    for (int i = 0; i < 4; ++i) {
        ra[i] = *(const float4*)(aptr[i]);
        rb[i] = *(const float4*)(bptr[i]);
    }

    for (int kk = 0; kk < D_MODEL; kk += 32) {
        __syncthreads();   // previous iteration's fragment reads complete

        // store prefetched chunk (fp32 -> bf16 hi/lo) into LDS
#pragma unroll
        for (int i = 0; i < 4; ++i) {
            store_split4(sAh, sAl, soff[i], ra[i]);
            store_split4(sBh, sBl, soff[i], rb[i]);
        }
        // issue next chunk's global loads; latency hidden by WMMA phase
        if (kk + 32 < D_MODEL) {
#pragma unroll
            for (int i = 0; i < 4; ++i) {
                ra[i] = *(const float4*)(aptr[i] + kk + 32);
                rb[i] = *(const float4*)(bptr[i] + kk + 32);
                if (kk + 64 < D_MODEL) __builtin_prefetch(bptr[i] + kk + 64, 0, 1);
            }
        }
        __syncthreads();

        // fragments + split-bf16 WMMA (full EXEC)
        v16bf ah[2], al[2], bgh[2], bgl[2], buh[2], bul[2];
#pragma unroll
        for (int t = 0; t < 2; ++t) {
            ah[t]  = load_frag_A(sAh, wr * 32 + t * 16, lane);
            al[t]  = load_frag_A(sAl, wr * 32 + t * 16, lane);
            bgh[t] = load_frag_B(sBh, wc * 32 + t * 16, lane);
            bgl[t] = load_frag_B(sBl, wc * 32 + t * 16, lane);
            buh[t] = load_frag_B(sBh, TFN + wc * 32 + t * 16, lane);
            bul[t] = load_frag_B(sBl, TFN + wc * 32 + t * 16, lane);
        }
#pragma unroll
        for (int tm = 0; tm < 2; ++tm)
#pragma unroll
            for (int tn = 0; tn < 2; ++tn) {
                accg[tm][tn] = wmma_bf16(ah[tm], bgh[tn], accg[tm][tn]);
                accg[tm][tn] = wmma_bf16(ah[tm], bgl[tn], accg[tm][tn]);
                accg[tm][tn] = wmma_bf16(al[tm], bgh[tn], accg[tm][tn]);
                accu[tm][tn] = wmma_bf16(ah[tm], buh[tn], accu[tm][tn]);
                accu[tm][tn] = wmma_bf16(ah[tm], bul[tn], accu[tm][tn]);
                accu[tm][tn] = wmma_bf16(al[tm], buh[tn], accu[tm][tn]);
            }
    }

    // ---- epilogue: silu(gate)*up -> bf16 hi/lo hidden ---------------------
    const int half = lane >> 4;
    const int n    = lane & 15;
#pragma unroll
    for (int tm = 0; tm < 2; ++tm)
#pragma unroll
        for (int tn = 0; tn < 2; ++tn)
#pragma unroll
            for (int r = 0; r < 8; ++r) {
                int mloc = wr * 32 + tm * 16 + half * 8 + r;
                int m = m_start + mloc;
                if (m < cnt) {
                    float g = accg[tm][tn][r];
                    float u = accu[tm][tn][r];
                    float h = (g / (1.0f + __expf(-g))) * u;
                    unsigned short hi, lo; split_bf16(h, hi, lo);
                    size_t off = (size_t)(base + m) * D_FF + (f0 + wc * 32 + tn * 16 + n);
                    hh[off] = hi;
                    hl[off] = lo;
                }
            }
}

// ---------------------------------------------------------------------------
// GEMM2:  out[tok, d] = hidden[p] · down[e][d]^T   (K = 2048, split-bf16)
// ---------------------------------------------------------------------------
__global__ __launch_bounds__(256)
void k_gemm2(const unsigned short* __restrict__ hh,
             const unsigned short* __restrict__ hl,
             const float* __restrict__ dwn,        // [8][D_MODEL][D_FF]
             const int* __restrict__ offsets,
             const int* __restrict__ perm,
             float* __restrict__ out) {
    const int e       = blockIdx.y;
    const int base    = offsets[e];
    const int cnt     = offsets[e + 1] - base;
    const int m_start = blockIdx.x * TM;
    if (m_start >= cnt) return;
    const int d0   = blockIdx.z * TFN;
    const int tid  = threadIdx.x;
    const int lane = tid & 31;
    const int w    = tid >> 5;
    const int wr   = w >> 1;
    const int wc   = w & 1;

    __shared__ unsigned short sAh[TM * LDA],  sAl[TM * LDA];
    __shared__ unsigned short sBh[TFN * LDA], sBl[TFN * LDA];
    __shared__ int sTok[TM];

    if (tid < TM) {
        int r = m_start + tid;
        sTok[tid] = perm[base + (r < cnt ? r : cnt - 1)];
    }

    // ---- hoisted per-thread staging pointers ------------------------------
    // A copy: thread t handles (row = t>>1, kc = (t&1)*16) : 16+16 shorts
    const int arow = tid >> 1;
    const int akc  = (tid & 1) * 16;
    {
        int r = m_start + arow;
        (void)r;
    }
    const size_t ap = (size_t)(base + ((m_start + arow) < cnt ? (m_start + arow) : cnt - 1));
    const uint4* aHp = (const uint4*)(hh + ap * D_FF + akc);
    const uint4* aLp = (const uint4*)(hl + ap * D_FF + akc);
    const int aoff = arow * LDA + akc;

    // B: 64x32 floats = 512 float4 slots, 2 per thread
    const float* wp = dwn + (size_t)e * D_MODEL * D_FF;
    const float* bptr[2];
    int boff[2];
#pragma unroll
    for (int i = 0; i < 2; ++i) {
        int linear = tid + i * 256;
        int r  = linear >> 3;
        int kq = (linear & 7) * 4;
        boff[i] = r * LDA + kq;
        bptr[i] = wp + (size_t)(d0 + r) * D_FF + kq;
    }

    v8f acc[2][2];
#pragma unroll
    for (int i = 0; i < 2; ++i)
#pragma unroll
        for (int j = 0; j < 2; ++j) acc[i][j] = vzero8();

    // ---- pipeline preload chunk 0 -----------------------------------------
    uint4 rh0, rh1, rl0, rl1;
    float4 rb[2];
    rh0 = aHp[0]; rh1 = aHp[1];
    rl0 = aLp[0]; rl1 = aLp[1];
#pragma unroll
    for (int i = 0; i < 2; ++i) rb[i] = *(const float4*)(bptr[i]);

    for (int kk = 0; kk < D_FF; kk += 32) {
        __syncthreads();

        // store prefetched chunk to LDS (A already bf16 hi/lo: raw copy)
        *(uint4*)&sAh[aoff]     = rh0;
        *(uint4*)&sAh[aoff + 8] = rh1;
        *(uint4*)&sAl[aoff]     = rl0;
        *(uint4*)&sAl[aoff + 8] = rl1;
#pragma unroll
        for (int i = 0; i < 2; ++i)
            store_split4(sBh, sBl, boff[i], rb[i]);

        // issue next chunk's loads
        if (kk + 32 < D_FF) {
            const uint4* nH = (const uint4*)((const unsigned short*)aHp + kk + 32);
            const uint4* nL = (const uint4*)((const unsigned short*)aLp + kk + 32);
            rh0 = nH[0]; rh1 = nH[1];
            rl0 = nL[0]; rl1 = nL[1];
#pragma unroll
            for (int i = 0; i < 2; ++i) {
                rb[i] = *(const float4*)(bptr[i] + kk + 32);
                if (kk + 64 < D_FF) __builtin_prefetch(bptr[i] + kk + 64, 0, 1);
            }
        }
        __syncthreads();

        v16bf ah[2], al[2], bh[2], bl[2];
#pragma unroll
        for (int t = 0; t < 2; ++t) {
            ah[t] = load_frag_A(sAh, wr * 32 + t * 16, lane);
            al[t] = load_frag_A(sAl, wr * 32 + t * 16, lane);
            bh[t] = load_frag_B(sBh, wc * 32 + t * 16, lane);
            bl[t] = load_frag_B(sBl, wc * 32 + t * 16, lane);
        }
#pragma unroll
        for (int tm = 0; tm < 2; ++tm)
#pragma unroll
            for (int tn = 0; tn < 2; ++tn) {
                acc[tm][tn] = wmma_bf16(ah[tm], bh[tn], acc[tm][tn]);
                acc[tm][tn] = wmma_bf16(ah[tm], bl[tn], acc[tm][tn]);
                acc[tm][tn] = wmma_bf16(al[tm], bh[tn], acc[tm][tn]);
            }
    }

    // ---- epilogue: scatter fp32 rows back to original token order ---------
    const int half = lane >> 4;
    const int n    = lane & 15;
#pragma unroll
    for (int tm = 0; tm < 2; ++tm)
#pragma unroll
        for (int tn = 0; tn < 2; ++tn)
#pragma unroll
            for (int r = 0; r < 8; ++r) {
                int mloc = wr * 32 + tm * 16 + half * 8 + r;
                int m = m_start + mloc;
                if (m < cnt) {
                    int tok = sTok[mloc];
                    out[(size_t)tok * D_MODEL + d0 + wc * 32 + tn * 16 + n] = acc[tm][tn][r];
                }
            }
}

// ---------------------------------------------------------------------------
// Launch
// ---------------------------------------------------------------------------
extern "C" void kernel_launch(void* const* d_in, const int* in_sizes, int n_in,
                              void* d_out, int out_size, void* d_ws, size_t ws_size,
                              hipStream_t stream) {
    const float* x   = (const float*)d_in[0];
    const int*   ei  = (const int*)d_in[1];
    const float* gup = (const float*)d_in[2];
    const float* dwn = (const float*)d_in[3];
    float*       out = (float*)d_out;
    (void)in_sizes; (void)n_in; (void)out_size; (void)ws_size;

    // ws: [counts 8][offsets 9][cursors][perm 16384] then hidden hi/lo bf16
    int* counts  = (int*)d_ws;
    int* offsets = counts + 8;
    int* cursors = counts + 32;
    int* perm    = counts + 64;                    // ends at byte 65792
    unsigned short* hh = (unsigned short*)((char*)d_ws + 65792);
    unsigned short* hl = hh + (size_t)NUM_TOKENS * D_FF;

    k_init   <<<1, 64, 0, stream>>>(counts, cursors);
    k_count  <<<NUM_TOKENS / 256, 256, 0, stream>>>(ei, counts);
    k_scan   <<<1, 32, 0, stream>>>(counts, offsets);
    k_scatter<<<NUM_TOKENS / 256, 256, 0, stream>>>(ei, offsets, cursors, perm);

    dim3 g1(NUM_TOKENS / TM, N_EXPERTS, D_FF / TFN);     // 128 x 8 x 32
    k_gemm1<<<g1, 256, 0, stream>>>(x, gup, offsets, perm, hh, hl);

    dim3 g2(NUM_TOKENS / TM, N_EXPERTS, D_MODEL / TFN);  // 128 x 8 x 16
    k_gemm2<<<g2, 256, 0, stream>>>(hh, hl, dwn, offsets, perm, out);
}
